// PrecedenceModule_36000415875165
// MI455X (gfx1250) — matrix operand order repaired and checked by reading, not compiled, not run on hardware
//
#include <hip/hip_runtime.h>
#include <hip/hip_bf16.h>

typedef _Float16 __attribute__((ext_vector_type(16))) v16h;
typedef _Float16 __attribute__((ext_vector_type(8)))  v8h;
typedef float    __attribute__((ext_vector_type(8)))  v8f;

static constexpr int kH    = 4;
static constexpr int kE    = 64;
static constexpr int kN    = 32;    // keys per row
static constexpr int kS    = 192;
static constexpr int kFF   = 256;
static constexpr int kROWS = 16;    // rows per block (16 | 192 -> no batch crossing)

// -------- workspace layout (bytes) --------
// [0,1024)       Wqk  f32 [16][16]   = Wq @ Wk^T
// [1024,9216)    WvoP f16 packed B-tiles of Wvo[64][64]  (folded Wv@Wo per head)
// [9216,41984)   W1P  f16 packed B-tiles of W1[64][256]
// [41984,74752)  W2P  f16 packed B-tiles of W2[256][64]
//
// Packed-B element order (per 32x16 K x N tile, contiguous K order, same as load_a_frag):
//   elem((kt,nt), lane, j) = W[kt*32 + (lane>>4)*16 + j][nt*16 + (lane&15)]
// so each lane loads its 16 halves with ONE contiguous 32-byte load.

__global__ __launch_bounds__(256) void prep_fold_pack(
    const float* __restrict__ Wq, const float* __restrict__ Wk,
    const float* __restrict__ Wv, const float* __restrict__ Wo,
    const float* __restrict__ W1, const float* __restrict__ W2,
    float* __restrict__ wqk, _Float16* __restrict__ wvoP,
    _Float16* __restrict__ w1P, _Float16* __restrict__ w2P)
{
    const int t = threadIdx.x;
    // Wqk[i][ip] = sum_j Wq[i][j] * Wk[ip][j]
    {
        int i = t >> 4, ip = t & 15;
        float acc = 0.f;
        #pragma unroll
        for (int j = 0; j < 16; ++j) acc += Wq[i * 16 + j] * Wk[ip * 16 + j];
        wqk[i * 16 + ip] = acc;
    }
    // Wvo packed: K=64, N=64 -> 2 x 4 tiles, 4096 halves.
    for (int p = t; p < 4096; p += 256) {
        int tile = p >> 9, lane = (p >> 4) & 31, j = p & 15;
        int kt = tile >> 2, nt = tile & 3;
        int krow = kt * 32 + ((lane >> 4) << 4) + j;   // 0..63 = h*16+d
        int col  = nt * 16 + (lane & 15);
        int h = krow >> 4, d = krow & 15;
        float acc = 0.f;
        #pragma unroll
        for (int k = 0; k < 16; ++k) acc += Wv[d * 16 + k] * Wo[(h * 16 + k) * 64 + col];
        wvoP[p] = (_Float16)acc;
    }
    // W1 packed: K=64, N=256 -> 2 x 16 tiles
    for (int p = t; p < 16384; p += 256) {
        int tile = p >> 9, lane = (p >> 4) & 31, j = p & 15;
        int kt = tile >> 4, nt = tile & 15;
        int krow = kt * 32 + ((lane >> 4) << 4) + j;
        int col  = nt * 16 + (lane & 15);
        w1P[p] = (_Float16)W1[krow * 256 + col];
    }
    // W2 packed: K=256, N=64 -> 8 x 4 tiles
    for (int p = t; p < 16384; p += 256) {
        int tile = p >> 9, lane = (p >> 4) & 31, j = p & 15;
        int kt = tile >> 2, nt = tile & 3;
        int krow = kt * 32 + ((lane >> 4) << 4) + j;
        int col  = nt * 16 + (lane & 15);
        w2P[p] = (_Float16)W2[krow * 64 + col];
    }
}

// A-fragment loader: contiguous K order (matches packed B).
// lanes 0-15 (hi=0): halves 0..15 = K 0..15
// lanes 16-31(hi=1): halves 0..15 = K 16..31
__device__ __forceinline__ v16h load_a_frag(const _Float16* __restrict__ rowp, int kt, int lhi)
{
    const v8h a0 = *(const v8h*)(rowp + kt * 32 + lhi * 16);
    const v8h a1 = *(const v8h*)(rowp + kt * 32 + lhi * 16 + 8);
    v16h a;
    #pragma unroll
    for (int i = 0; i < 8; ++i) { a[i] = a0[i]; a[8 + i] = a1[i]; }
    return a;
}

__global__ __launch_bounds__(128) void precedence_block_kernel(
    const float* __restrict__ prec, const int* __restrict__ top_mask,
    const float* __restrict__ We,  const float* __restrict__ be,
    const float* __restrict__ bo,
    const float* __restrict__ g1,  const float* __restrict__ b1ln,
    const float* __restrict__ b1,  const float* __restrict__ b2,
    const float* __restrict__ g2,  const float* __restrict__ b2ln,
    const float* __restrict__ wqk, const _Float16* __restrict__ wvoP,
    const _Float16* __restrict__ w1P, const _Float16* __restrict__ w2P,
    float* __restrict__ out)
{
    __shared__ float prec_s[kROWS][66];            // padded: conflict-free row access
    __shared__ float msk_s[kN];
    __shared__ float wqk_s[16][17];
    __shared__ float q_s[kROWS][68];               // q (residual input), f32
    __shared__ float S0_s[kROWS][kH], S1_s[kROWS][kH];
    __shared__ __align__(16) _Float16 a16[kROWS][72];    // ws-f16 then x-f16 (reused)
    __shared__ float xbuf[kROWS][68];              // ao+q -> normalized x
    __shared__ __align__(16) _Float16 hbuf[kROWS][264];  // relu(xW1+b1), f16
    __shared__ float ybuf[kROWS][68];              // ff + x
    __shared__ float mu_s[kROWS], rs_s[kROWS];

    const int t   = threadIdx.x;
    const int m0  = blockIdx.x * kROWS;
    const int bat = m0 / kS;                       // uniform per block

    // ---- stage precedence tile (rows m0..m0+15, all 32 keys x 2) : 4KB, coalesced
    {
        const float* src = prec + (size_t)m0 * 64;
        #pragma unroll
        for (int k = 0; k < 8; ++k) {
            int idx = k * 128 + t;                 // 0..1023
            prec_s[idx >> 6][idx & 63] = src[idx];
        }
    }
    if (t < kN) msk_s[t] = (float)top_mask[bat * kN + t];
    for (int p = t; p < 256; p += 128) wqk_s[p >> 4][p & 15] = wqk[p];
    __syncthreads();

    // ---- q[m,e] = p0q*We0[e] + p1q*We1[e] + be[e]   (pv gathered at n = s % 32)
    {
        int r = t >> 3, c = t & 7;
        int s = (m0 + r) % kS;
        int nq = s & (kN - 1);
        float p0q = prec_s[r][2 * nq], p1q = prec_s[r][2 * nq + 1];
        #pragma unroll
        for (int k = 0; k < 8; ++k) {
            int e = c * 8 + k;
            q_s[r][e] = p0q * We[e] + p1q * We[64 + e] + be[e];
        }
    }
    __syncthreads();

    // ---- attention, fully collapsed: per (row, head) thread
    if (t < 64) {
        int r = t >> 2, h = t & 3;
        float qt[16];
        #pragma unroll
        for (int d2 = 0; d2 < 16; ++d2) {
            float a = 0.f;
            #pragma unroll
            for (int d = 0; d < 16; ++d) a += q_s[r][h * 16 + d] * wqk_s[d][d2];
            qt[d2] = a;
        }
        float Ah = 0.f, Bh = 0.f, Ch = 0.f;
        #pragma unroll
        for (int d2 = 0; d2 < 16; ++d2) {
            Ah += qt[d2] * We[h * 16 + d2];
            Bh += qt[d2] * We[64 + h * 16 + d2];
            Ch += qt[d2] * be[h * 16 + d2];
        }
        float z[kN]; float zmax = -3.0e38f;
        #pragma unroll
        for (int n = 0; n < kN; ++n) {
            float p0 = prec_s[r][2 * n], p1 = prec_s[r][2 * n + 1];
            float zz = (msk_s[n] == 0.f) ? -1.0e30f
                                         : (Ah * p0 + Bh * p1 + Ch) * 0.125f;
            z[n] = zz; zmax = fmaxf(zmax, zz);
        }
        float sum = 0.f, s0 = 0.f, s1 = 0.f;
        #pragma unroll
        for (int n = 0; n < kN; ++n) {
            float p = __expf(z[n] - zmax);
            sum += p;
            s0 += p * prec_s[r][2 * n];
            s1 += p * prec_s[r][2 * n + 1];
        }
        float inv = 1.f / sum;
        S0_s[r][h] = s0 * inv; S1_s[r][h] = s1 * inv;
    }
    __syncthreads();

    // ---- ws[e] = S0_h*We0[e] + S1_h*We1[e] + be[e]  -> f16 A-tile
    {
        int r = t >> 3, c = t & 7;
        #pragma unroll
        for (int k = 0; k < 8; ++k) {
            int e = c * 8 + k; int h = e >> 4;
            float wv = S0_s[r][h] * We[e] + S1_s[r][h] * We[64 + e] + be[e];
            a16[r][e] = (_Float16)wv;
        }
    }
    __syncthreads();

    const int wv_  = t >> 5;        // wave id: owns output cols wv_*16..+15
    const int lane = t & 31;
    const int lrow = lane & 15;
    const int lhi  = lane >> 4;

    // ---- GEMM1: ao = ws @ Wvo  (16x64 @ 64x64), epilogue: +bo +q
    {
        v8f c0 = {};
        #pragma unroll
        for (int kt = 0; kt < 2; ++kt) {
            v16h a  = load_a_frag(&a16[lrow][0], kt, lhi);
            v16h bm = *(const v16h*)(wvoP + (size_t)((kt * 4 + wv_) * 32 + lane) * 16);
            c0 = __builtin_amdgcn_wmma_f32_16x16x32_f16(false, a, false, bm,
                                                        (short)0, c0, false, false);
        }
        #pragma unroll
        for (int v = 0; v < 8; ++v) {
            int row = lhi ? v + 8 : v;
            int col = wv_ * 16 + lrow;
            xbuf[row][col] = c0[v] + bo[col] + q_s[row][col];
        }
    }
    __syncthreads();

    // ---- LayerNorm 1
    if (t < kROWS) {
        float s = 0.f, s2 = 0.f;
        for (int e = 0; e < kE; ++e) { float x = xbuf[t][e]; s += x; s2 += x * x; }
        float mu = s * (1.f / 64.f);
        float var = s2 * (1.f / 64.f) - mu * mu;
        mu_s[t] = mu; rs_s[t] = rsqrtf(var + 1e-5f);
    }
    __syncthreads();
    {
        int r = t >> 3, c = t & 7;
        float mu = mu_s[r], rs = rs_s[r];
        #pragma unroll
        for (int k = 0; k < 8; ++k) {
            int e = c * 8 + k;
            float xv = (xbuf[r][e] - mu) * rs * g1[e] + b1ln[e];
            xbuf[r][e] = xv;                 // keep f32 for residual
            a16[r][e]  = (_Float16)xv;       // A-tile for GEMM2
        }
    }
    __syncthreads();

    // ---- GEMM2: h = relu(x @ W1 + b1)  (16x64 @ 64x256); 4 N-tiles per wave
    {
        v16h a0 = load_a_frag(&a16[lrow][0], 0, lhi);
        v16h a1 = load_a_frag(&a16[lrow][0], 1, lhi);
        #pragma unroll
        for (int sub = 0; sub < 4; ++sub) {
            int nt = wv_ * 4 + sub;
            v8f c = {};
            v16h b0 = *(const v16h*)(w1P + (size_t)((nt) * 32 + lane) * 16);
            c = __builtin_amdgcn_wmma_f32_16x16x32_f16(false, a0, false, b0,
                                                       (short)0, c, false, false);
            v16h bx = *(const v16h*)(w1P + (size_t)((16 + nt) * 32 + lane) * 16);
            c = __builtin_amdgcn_wmma_f32_16x16x32_f16(false, a1, false, bx,
                                                       (short)0, c, false, false);
            #pragma unroll
            for (int v = 0; v < 8; ++v) {
                int row = lhi ? v + 8 : v;
                int col = nt * 16 + lrow;
                float hv = c[v] + b1[col];
                hbuf[row][col] = (_Float16)(hv > 0.f ? hv : 0.f);
            }
        }
    }
    __syncthreads();

    // ---- GEMM3: ff = h @ W2 + b2  (16x256 @ 256x64), epilogue: + x residual
    {
        v8f c = {};
        #pragma unroll
        for (int kt = 0; kt < 8; ++kt) {
            v16h a  = load_a_frag(&hbuf[lrow][0], kt, lhi);
            v16h bm = *(const v16h*)(w2P + (size_t)((kt * 4 + wv_) * 32 + lane) * 16);
            c = __builtin_amdgcn_wmma_f32_16x16x32_f16(false, a, false, bm,
                                                       (short)0, c, false, false);
        }
        #pragma unroll
        for (int v = 0; v < 8; ++v) {
            int row = lhi ? v + 8 : v;
            int col = wv_ * 16 + lrow;
            ybuf[row][col] = c[v] + b2[col] + xbuf[row][col];
        }
    }
    __syncthreads();

    // ---- LayerNorm 2 + store
    if (t < kROWS) {
        float s = 0.f, s2 = 0.f;
        for (int e = 0; e < kE; ++e) { float x = ybuf[t][e]; s += x; s2 += x * x; }
        float mu = s * (1.f / 64.f);
        float var = s2 * (1.f / 64.f) - mu * mu;
        mu_s[t] = mu; rs_s[t] = rsqrtf(var + 1e-5f);
    }
    __syncthreads();
    {
        int r = t >> 3, c = t & 7;
        float mu = mu_s[r], rs = rs_s[r];
        float* op = out + (size_t)(m0 + r) * kE;
        #pragma unroll
        for (int k = 0; k < 8; ++k) {
            int e = c * 8 + k;
            op[e] = (ybuf[r][e] - mu) * rs * g2[e] + b2ln[e];
        }
    }
}

extern "C" void kernel_launch(void* const* d_in, const int* in_sizes, int n_in,
                              void* d_out, int out_size, void* d_ws, size_t ws_size,
                              hipStream_t stream)
{
    const float* prec = (const float*)d_in[0];
    const int*   tm   = (const int*)  d_in[1];
    const float* We   = (const float*)d_in[2];
    const float* be   = (const float*)d_in[3];
    const float* Wq   = (const float*)d_in[4];
    const float* Wk   = (const float*)d_in[5];
    const float* Wv   = (const float*)d_in[6];
    const float* Wo   = (const float*)d_in[7];
    const float* bo   = (const float*)d_in[8];
    const float* g1   = (const float*)d_in[9];
    const float* b1l  = (const float*)d_in[10];
    const float* W1   = (const float*)d_in[11];
    const float* b1   = (const float*)d_in[12];
    const float* W2   = (const float*)d_in[13];
    const float* b2   = (const float*)d_in[14];
    const float* g2   = (const float*)d_in[15];
    const float* b2l  = (const float*)d_in[16];

    char* ws = (char*)d_ws;
    float*    wqk  = (float*)ws;
    _Float16* wvoP = (_Float16*)(ws + 1024);
    _Float16* w1P  = (_Float16*)(ws + 9216);
    _Float16* w2P  = (_Float16*)(ws + 41984);

    prep_fold_pack<<<1, 256, 0, stream>>>(Wq, Wk, Wv, Wo, W1, W2, wqk, wvoP, w1P, w2P);

    const int M = in_sizes[0] / (kN * 2);   // B*S = 24576
    precedence_block_kernel<<<M / kROWS, 128, 0, stream>>>(
        prec, tm, We, be, bo, g1, b1l, b1, b2, g2, b2l,
        wqk, wvoP, w1P, w2P, (float*)d_out);
}
